// DiffusionConditioning_8624294330887
// MI455X (gfx1250) — matrix operand order repaired and checked by reading, not compiled, not run on hardware
//
#include <hip/hip_runtime.h>
#include <hip/hip_fp16.h>

typedef __attribute__((ext_vector_type(16))) _Float16 v16h;
typedef __attribute__((ext_vector_type(8)))  float    v8f;

// ---------------------------------------------------------------------------
// WMMA helpers (gfx1250, wave32).  D(16x16,f32) = A(16x32,f16) x B(32x16,f16) + C
// A layout (ISA 7.12.2, 16-bit A 16x32): lanes 0-15 -> row M=lane, K {0-7,16-23};
//                                        lanes 16-31 -> row M=lane-16, K {8-15,24-31}
// B layout (same K order as A):          lanes 0-15 -> col N=lane, K {0-7,16-23};
//                                        lanes 16-31 -> col N=lane-16, K {8-15,24-31}
// C/D layout: lane%16 = col, vgpr e -> row e (+8 for lanes 16-31)
// ---------------------------------------------------------------------------
__device__ __forceinline__ v8f wmma16(v16h a, v16h b, v8f c) {
  return __builtin_amdgcn_wmma_f32_16x16x32_f16(false, a, false, b, (short)0, c,
                                                false, false);
}

__device__ __forceinline__ v16h load_a_lds(const _Float16* buf, int stride, int k0,
                                           int lane) {
  int r  = lane & 15;
  int ko = k0 + ((lane & 16) ? 8 : 0);
  const _Float16* p = buf + r * stride + ko;
  union { v16h v; uint4 q[2]; } A;
  A.q[0] = *(const uint4*)(p);
  A.q[1] = *(const uint4*)(p + 16);
  return A.v;
}

__device__ __forceinline__ v16h load_b_glb(const _Float16* wT, int stride, int n0,
                                           int k0, int lane) {
  int n  = n0 + (lane & 15);
  int ko = k0 + ((lane & 16) ? 8 : 0);
  const _Float16* p = wT + n * stride + ko;
  union { v16h v; uint4 q[2]; } B;
  B.q[0] = *(const uint4*)(p);
  B.q[1] = *(const uint4*)(p + 16);
  return B.v;
}

__device__ __forceinline__ float silu_f(float b) {
  return b / (1.0f + __expf(-b));
}

// ---------------------------------------------------------------------------
// Weight prep: f32 [K][N] row-major -> f16 [N][KP] (N-major, K zero-padded)
// ---------------------------------------------------------------------------
__global__ void k_transpose_f16(const float* __restrict__ in, _Float16* __restrict__ out,
                                int K, int N, int KP) {
  int total = N * KP;
  for (int idx = blockIdx.x * blockDim.x + threadIdx.x; idx < total;
       idx += gridDim.x * blockDim.x) {
    int n = idx / KP, k = idx - n * KP;
    out[idx] = (k < K) ? (_Float16)in[k * N + n] : (_Float16)0.0f;
  }
}

// ---------------------------------------------------------------------------
// Noise embedding: n[m][j] = LN(cos(2pi(t*fw+fb)), g) @ w_noise   (4 x 384)
// ---------------------------------------------------------------------------
__global__ void k_noise_embed(const float* __restrict__ noise,
                              const float* __restrict__ fw,
                              const float* __restrict__ fb,
                              const float* __restrict__ g,
                              const float* __restrict__ w_noise,
                              float* __restrict__ n_out) {
  __shared__ float fe[256];
  __shared__ float red[256];
  int tid = threadIdx.x;
  for (int m = 0; m < 4; ++m) {
    float t = logf(noise[m] * (1.0f / 16.0f)) * 0.25f;
    float v = cosf(6.28318530717958647692f * (t * fw[tid] + fb[tid]));
    red[tid] = v;
    __syncthreads();
    for (int s = 128; s > 0; s >>= 1) {
      if (tid < s) red[tid] += red[tid + s];
      __syncthreads();
    }
    float mean = red[0] * (1.0f / 256.0f);
    __syncthreads();
    float d = v - mean;
    red[tid] = d * d;
    __syncthreads();
    for (int s = 128; s > 0; s >>= 1) {
      if (tid < s) red[tid] += red[tid + s];
      __syncthreads();
    }
    float inv = rsqrtf(red[0] * (1.0f / 256.0f) + 1e-5f);
    __syncthreads();
    fe[tid] = d * inv * g[tid];
    __syncthreads();
    for (int j = tid; j < 384; j += 256) {
      float acc = 0.0f;
      for (int k = 0; k < 256; ++k) acc += fe[k] * w_noise[k * 384 + j];
      n_out[m * 384 + j] = acc;
    }
    __syncthreads();
  }
}

// ---------------------------------------------------------------------------
// s0 = LN(concat(trunk_single_cond, inputs), g) @ w_single   (768 x 384)
// 48 blocks x 16 rows, 8 waves, K=833 padded to 864 (27 K-chunks), N-tiles 24
// ---------------------------------------------------------------------------
__global__ void k_single_s0(const float* __restrict__ tsc,
                            const float* __restrict__ inp,
                            const float* __restrict__ g,
                            const _Float16* __restrict__ wT,  // [384][864]
                            float* __restrict__ s0) {
  const int KP = 864, KC = 833;
  __shared__ __align__(16) _Float16 lnbuf[16 * 864];
  __shared__ float rs[256], rq[256], rmean[16], rinv[16];
  int tid = threadIdx.x;
  int r = tid >> 4, p = tid & 15;
  int l = blockIdx.x * 16 + r;

  float s = 0.0f, ss = 0.0f;
  for (int c = p; c < KC; c += 16) {
    float v = (c < 384) ? tsc[l * 384 + c] : inp[l * 449 + (c - 384)];
    s += v; ss += v * v;
  }
  rs[tid] = s; rq[tid] = ss;
  __syncthreads();
  if (tid < 16) {
    float a = 0.0f, b = 0.0f;
    for (int q = 0; q < 16; ++q) { a += rs[tid * 16 + q]; b += rq[tid * 16 + q]; }
    float mean = a / 833.0f;
    rmean[tid] = mean;
    rinv[tid] = rsqrtf(b / 833.0f - mean * mean + 1e-5f);
  }
  __syncthreads();
  float mean = rmean[r], inv = rinv[r];
  for (int c = p; c < KC; c += 16) {
    float v = (c < 384) ? tsc[l * 384 + c] : inp[l * 449 + (c - 384)];
    lnbuf[r * KP + c] = (_Float16)((v - mean) * inv * g[c]);
  }
  for (int c = KC + p; c < KP; c += 16) lnbuf[r * KP + c] = (_Float16)0.0f;
  __syncthreads();

  int lane = tid & 31, wv = tid >> 5;
  int col = lane & 15, rbase = (lane & 16) >> 1;
  v8f acc[3] = {};
  for (int kc = 0; kc < 27; ++kc) {
    v16h A = load_a_lds(lnbuf, KP, kc * 32, lane);
    for (int t = 0; t < 3; ++t) {
      int n0 = (wv * 3 + t) * 16;
      v16h B = load_b_glb(wT, KP, n0, kc * 32, lane);
      acc[t] = wmma16(A, B, acc[t]);
    }
  }
  for (int t = 0; t < 3; ++t) {
    int n0 = (wv * 3 + t) * 16;
    for (int e = 0; e < 8; ++e)
      s0[(blockIdx.x * 16 + rbase + e) * 384 + n0 + col] = acc[t][e];
  }
}

// ---------------------------------------------------------------------------
// Single-stream SwiGLU FFN (D=384, hidden 1536, rows staged in LDS)
// ---------------------------------------------------------------------------
__device__ void ffn_single(float* xres, _Float16* lnbuf, _Float16* gbuf,
                           float* rs, float* rq, float* rmean, float* rinv,
                           const float* __restrict__ g,
                           const _Float16* __restrict__ wiT,   // [1536][384]
                           const _Float16* __restrict__ woT,   // [384][768]
                           int tid) {
  int r = tid >> 4, p = tid & 15;
  float s = 0.0f, ss = 0.0f;
  for (int c = p; c < 384; c += 16) { float v = xres[r * 384 + c]; s += v; ss += v * v; }
  rs[tid] = s; rq[tid] = ss;
  __syncthreads();
  if (tid < 16) {
    float a = 0.0f, b = 0.0f;
    for (int q = 0; q < 16; ++q) { a += rs[tid * 16 + q]; b += rq[tid * 16 + q]; }
    float mean = a / 384.0f;
    rmean[tid] = mean;
    rinv[tid] = rsqrtf(b / 384.0f - mean * mean + 1e-5f);
  }
  __syncthreads();
  float mean = rmean[r], inv = rinv[r];
  for (int c = p; c < 384; c += 16)
    lnbuf[r * 384 + c] = (_Float16)((xres[r * 384 + c] - mean) * inv * g[c]);
  __syncthreads();

  int lane = tid & 31, wv = tid >> 5;
  int col = lane & 15, rbase = (lane & 16) >> 1;
  // w_in + fused SwiGLU: 48 (a,b) tile pairs, 6 per wave
  for (int pr = 0; pr < 6; ++pr) {
    int at = wv * 6 + pr;
    int na = at * 16, nb = 768 + at * 16;
    v8f accA = {}, accB = {};
    for (int kc = 0; kc < 12; ++kc) {
      v16h A  = load_a_lds(lnbuf, 384, kc * 32, lane);
      v16h Ba = load_b_glb(wiT, 384, na, kc * 32, lane);
      v16h Bb = load_b_glb(wiT, 384, nb, kc * 32, lane);
      accA = wmma16(A, Ba, accA);
      accB = wmma16(A, Bb, accB);
    }
    for (int e = 0; e < 8; ++e)
      gbuf[(rbase + e) * 768 + na + col] = (_Float16)(accA[e] * silu_f(accB[e]));
  }
  __syncthreads();
  // w_out: 24 N-tiles, 3 per wave, K=768
  v8f acc[3] = {};
  for (int kc = 0; kc < 24; ++kc) {
    v16h A = load_a_lds(gbuf, 768, kc * 32, lane);
    for (int t = 0; t < 3; ++t) {
      int n0 = (wv * 3 + t) * 16;
      v16h B = load_b_glb(woT, 768, n0, kc * 32, lane);
      acc[t] = wmma16(A, B, acc[t]);
    }
  }
  for (int t = 0; t < 3; ++t) {
    int n0 = (wv * 3 + t) * 16;
    for (int e = 0; e < 8; ++e)
      xres[(rbase + e) * 384 + n0 + col] += acc[t][e];
  }
  __syncthreads();
}

// rows flattened (m,l): 192 blocks x 16 rows
__global__ void k_single_ffn(const float* __restrict__ s0,
                             const float* __restrict__ nbuf,
                             const float* __restrict__ g1,
                             const _Float16* __restrict__ wi1T,
                             const _Float16* __restrict__ wo1T,
                             const float* __restrict__ g2,
                             const _Float16* __restrict__ wi2T,
                             const _Float16* __restrict__ wo2T,
                             float* __restrict__ out_s) {
  __shared__ __align__(16) _Float16 lnbuf[16 * 384];
  __shared__ __align__(16) _Float16 gbuf[16 * 768];
  __shared__ float xres[16 * 384];
  __shared__ float rs[256], rq[256], rmean[16], rinv[16];
  int tid = threadIdx.x;
  int R0 = blockIdx.x * 16;
  int m = R0 / 768, l0 = R0 % 768;

  for (int idx = tid; idx < 16 * 384; idx += 256) {
    int rr = idx / 384, cc = idx - rr * 384;
    xres[idx] = s0[(l0 + rr) * 384 + cc] + nbuf[m * 384 + cc];
  }
  __syncthreads();
  ffn_single(xres, lnbuf, gbuf, rs, rq, rmean, rinv, g1, wi1T, wo1T, tid);
  ffn_single(xres, lnbuf, gbuf, rs, rq, rmean, rinv, g2, wi2T, wo2T, tid);
  for (int idx = tid; idx < 16 * 384; idx += 256)
    out_s[(size_t)R0 * 384 + idx] = xres[idx];
}

// ---------------------------------------------------------------------------
// Pair-stream SwiGLU FFN, 32 rows (2 x 16-row A-subtiles share each B frag)
// All B fragments for each K-loop are hoisted into registers: one global
// clause + one wait per phase instead of per-WMMA waits.
// ---------------------------------------------------------------------------
__device__ void ffn_pair32(float* xres, _Float16* abuf, _Float16* gbuf,
                           float* rs, float* rq, float* rmean, float* rinv,
                           const float* __restrict__ g,
                           const _Float16* __restrict__ wiT,   // [512][128]
                           const _Float16* __restrict__ woT,   // [128][256]
                           int tid) {
  int r = tid >> 3, p = tid & 7;   // 32 rows x 8 threads
  float s = 0.0f, ss = 0.0f;
  for (int c = p; c < 128; c += 8) { float v = xres[r * 128 + c]; s += v; ss += v * v; }
  rs[tid] = s; rq[tid] = ss;
  __syncthreads();
  if (tid < 32) {
    float a = 0.0f, b = 0.0f;
    for (int q = 0; q < 8; ++q) { a += rs[tid * 8 + q]; b += rq[tid * 8 + q]; }
    float mean = a / 128.0f;
    rmean[tid] = mean;
    rinv[tid] = rsqrtf(b / 128.0f - mean * mean + 1e-5f);
  }
  __syncthreads();
  float mean = rmean[r], inv = rinv[r];
  for (int c = p; c < 128; c += 8)
    abuf[r * 256 + c] = (_Float16)((xres[r * 128 + c] - mean) * inv * g[c]);
  __syncthreads();

  int lane = tid & 31, wv = tid >> 5;
  int col = lane & 15, rbase = (lane & 16) >> 1;
  const _Float16* abuf1 = abuf + 16 * 256;
  // w_in + fused SwiGLU: 16 (a,b) tile pairs, 2 per wave, K=128, 2 subtiles
  for (int pr = 0; pr < 2; ++pr) {
    int at = wv * 2 + pr;
    int na = at * 16, nb = 256 + at * 16;
    v16h Ba[4], Bb[4];
#pragma unroll
    for (int kc = 0; kc < 4; ++kc) {
      Ba[kc] = load_b_glb(wiT, 128, na, kc * 32, lane);
      Bb[kc] = load_b_glb(wiT, 128, nb, kc * 32, lane);
    }
    v8f aA0 = {}, aA1 = {}, aB0 = {}, aB1 = {};
#pragma unroll
    for (int kc = 0; kc < 4; ++kc) {
      v16h A0 = load_a_lds(abuf, 256, kc * 32, lane);
      v16h A1 = load_a_lds(abuf1, 256, kc * 32, lane);
      aA0 = wmma16(A0, Ba[kc], aA0);
      aA1 = wmma16(A1, Ba[kc], aA1);
      aB0 = wmma16(A0, Bb[kc], aB0);
      aB1 = wmma16(A1, Bb[kc], aB1);
    }
    for (int e = 0; e < 8; ++e) {
      gbuf[(rbase + e) * 256 + na + col]      = (_Float16)(aA0[e] * silu_f(aB0[e]));
      gbuf[(16 + rbase + e) * 256 + na + col] = (_Float16)(aA1[e] * silu_f(aB1[e]));
    }
  }
  __syncthreads();
  // w_out: 8 N-tiles, 1 per wave, K=256, 2 subtiles, B hoisted
  int n0 = wv * 16;
  v16h Bo[8];
#pragma unroll
  for (int kc = 0; kc < 8; ++kc)
    Bo[kc] = load_b_glb(woT, 256, n0, kc * 32, lane);
  v8f o0 = {}, o1 = {};
  const _Float16* gbuf1 = gbuf + 16 * 256;
#pragma unroll
  for (int kc = 0; kc < 8; ++kc) {
    v16h A0 = load_a_lds(gbuf, 256, kc * 32, lane);
    v16h A1 = load_a_lds(gbuf1, 256, kc * 32, lane);
    o0 = wmma16(A0, Bo[kc], o0);
    o1 = wmma16(A1, Bo[kc], o1);
  }
  for (int e = 0; e < 8; ++e) {
    xres[(rbase + e) * 128 + n0 + col]        += o0[e];
    xres[(16 + rbase + e) * 128 + n0 + col]   += o1[e];
  }
  __syncthreads();
}

// Fused pair pipeline: relpos gather -> LN(concat) -> @w_pair -> 2x FFN -> out
// grid = 768*24 blocks; block = fixed i, 32 consecutive j
__global__ void k_pair(const int* __restrict__ sidx, const int* __restrict__ scol,
                       const int* __restrict__ ssym, const int* __restrict__ sent,
                       const float* __restrict__ tpc, const float* __restrict__ wrel,
                       const float* __restrict__ lnpg,
                       const _Float16* __restrict__ wpT,     // [128][256]
                       const float* __restrict__ pg1,
                       const _Float16* __restrict__ pwi1T,
                       const _Float16* __restrict__ pwo1T,
                       const float* __restrict__ pg2,
                       const _Float16* __restrict__ pwi2T,
                       const _Float16* __restrict__ pwo2T,
                       float* __restrict__ out_x) {
  __shared__ __align__(16) _Float16 abuf[32 * 256];   // 16 KB (LN output, f16)
  __shared__ __align__(16) float    catbuf[32 * 256]; // 32 KB concat staging; gbuf overlays
  __shared__ float xres[32 * 128];                    // 16 KB residual
  __shared__ float rs[256], rq[256], rmean[32], rinv[32];
  __shared__ int s_i1[32], s_i2[32], s_i3[32], s_be[32];
  _Float16* gbuf = (_Float16*)catbuf;                 // 16 KB inside dead catbuf

  int tid = threadIdx.x;
  int bi = blockIdx.x;
  int i = bi / 24;
  int j0 = (bi - i * 24) * 32;

  if (tid < 32) {
    int j = j0 + tid;
    int si = sidx[i], sj = sidx[j];
    int di = si - sj;
    int dic = (di < -32 ? -32 : (di > 32 ? 32 : di)) + 32;
    bool bcol = (scol[i] == scol[j]);
    bool bidx = (si == sj);
    bool bent = (sent[i] == sent[j]);
    int ds = ssym[i] - ssym[j];
    int dsc = (ds < -2 ? -2 : (ds > 2 ? 2 : ds)) + 2;
    s_i1[tid] = bcol ? dic : 65;
    s_i2[tid] = (bcol && bidx) ? dic : 65;
    s_i3[tid] = bent ? dsc : 5;
    s_be[tid] = bent ? 1 : 0;
  }
  // stage trunk_pair_cond into catbuf[:, 0:128]
  for (int idx = tid; idx < 32 * 128; idx += 256) {
    int rr = idx >> 7, cc = idx & 127;
    catbuf[rr * 256 + cc] = tpc[((size_t)(i * 768 + j0 + rr)) * 128 + cc];
  }
  __syncthreads();
  // relpos gather into catbuf[:, 128:256]
  for (int idx = tid; idx < 32 * 128; idx += 256) {
    int rr = idx >> 7, cc = idx & 127;
    float v = wrel[s_i1[rr] * 128 + cc] + wrel[(66 + s_i2[rr]) * 128 + cc] +
              (s_be[rr] ? wrel[132 * 128 + cc] : 0.0f) +
              wrel[(133 + s_i3[rr]) * 128 + cc];
    catbuf[rr * 256 + 128 + cc] = v;
  }
  __syncthreads();

  // LN over concat (256-wide), 8 threads per row
  int r = tid >> 3, p = tid & 7;
  float s = 0.0f, ss = 0.0f;
  for (int c = p; c < 256; c += 8) {
    float v = catbuf[r * 256 + c];
    s += v; ss += v * v;
  }
  rs[tid] = s; rq[tid] = ss;
  __syncthreads();
  if (tid < 32) {
    float a = 0.0f, b = 0.0f;
    for (int q = 0; q < 8; ++q) { a += rs[tid * 8 + q]; b += rq[tid * 8 + q]; }
    float mean = a / 256.0f;
    rmean[tid] = mean;
    rinv[tid] = rsqrtf(b / 256.0f - mean * mean + 1e-5f);
  }
  __syncthreads();
  {
    float mean = rmean[r], inv = rinv[r];
    for (int c = p; c < 256; c += 8)
      abuf[r * 256 + c] = (_Float16)((catbuf[r * 256 + c] - mean) * inv * lnpg[c]);
  }
  __syncthreads();

  // x0 = abuf(32x256) @ w_pair (256x128): 8 N-tiles, 1 per wave, 2 subtiles
  int lane = tid & 31, wv = tid >> 5;
  int col = lane & 15, rbase = (lane & 16) >> 1;
  {
    int n0 = wv * 16;
    v16h Bp[8];
#pragma unroll
    for (int kc = 0; kc < 8; ++kc)
      Bp[kc] = load_b_glb(wpT, 256, n0, kc * 32, lane);
    v8f a0 = {}, a1 = {};
    const _Float16* abuf1 = abuf + 16 * 256;
#pragma unroll
    for (int kc = 0; kc < 8; ++kc) {
      v16h A0 = load_a_lds(abuf, 256, kc * 32, lane);
      v16h A1 = load_a_lds(abuf1, 256, kc * 32, lane);
      a0 = wmma16(A0, Bp[kc], a0);
      a1 = wmma16(A1, Bp[kc], a1);
    }
    for (int e = 0; e < 8; ++e) {
      xres[(rbase + e) * 128 + n0 + col]      = a0[e];
      xres[(16 + rbase + e) * 128 + n0 + col] = a1[e];
    }
  }
  __syncthreads();

  ffn_pair32(xres, abuf, gbuf, rs, rq, rmean, rinv, pg1, pwi1T, pwo1T, tid);
  ffn_pair32(xres, abuf, gbuf, rs, rq, rmean, rinv, pg2, pwi2T, pwo2T, tid);

  size_t base = ((size_t)(i * 768 + j0)) * 128;
  for (int idx = tid; idx < 32 * 128; idx += 256)
    out_x[base + idx] = xres[idx];
}

// ---------------------------------------------------------------------------
// Launch
// ---------------------------------------------------------------------------
extern "C" void kernel_launch(void* const* d_in, const int* in_sizes, int n_in,
                              void* d_out, int out_size, void* d_ws, size_t ws_size,
                              hipStream_t stream) {
  (void)in_sizes; (void)n_in; (void)out_size; (void)ws_size;

  // inputs (setup_inputs order; params leaves in dict/insertion order)
  const int*   seq_index = (const int*)d_in[0];
  const int*   seq_color = (const int*)d_in[1];
  const int*   seq_sym   = (const int*)d_in[2];
  const int*   seq_entity= (const int*)d_in[3];
  const float* noise     = (const float*)d_in[4];
  const float* inputs    = (const float*)d_in[5];
  const float* tsc       = (const float*)d_in[6];
  const float* tpc       = (const float*)d_in[7];
  const float* ln_single_g = (const float*)d_in[8];
  const float* w_single    = (const float*)d_in[9];
  const float* fourier_w   = (const float*)d_in[10];
  const float* fourier_b   = (const float*)d_in[11];
  const float* ln_noise_g  = (const float*)d_in[12];
  const float* w_noise     = (const float*)d_in[13];
  const float* w_relpos    = (const float*)d_in[14];
  const float* ln_pair_g   = (const float*)d_in[15];
  const float* w_pair      = (const float*)d_in[16];
  const float* sg0  = (const float*)d_in[17];
  const float* swi0 = (const float*)d_in[18];
  const float* swo0 = (const float*)d_in[19];
  const float* sg1  = (const float*)d_in[20];
  const float* swi1 = (const float*)d_in[21];
  const float* swo1 = (const float*)d_in[22];
  const float* pg0  = (const float*)d_in[23];
  const float* pwi0 = (const float*)d_in[24];
  const float* pwo0 = (const float*)d_in[25];
  const float* pg1  = (const float*)d_in[26];
  const float* pwi1 = (const float*)d_in[27];
  const float* pwo1 = (const float*)d_in[28];

  // workspace layout (16B-aligned regions)
  char* w = (char*)d_ws;
  size_t off = 0;
  float* n_buf = (float*)(w + off);        off += 4 * 384 * 4;      // 6144
  float* s0    = (float*)(w + off);        off += 768 * 384 * 4;    // 1179648
  _Float16* wsT   = (_Float16*)(w + off);  off += (size_t)384 * 864 * 2;
  _Float16* swi0T = (_Float16*)(w + off);  off += (size_t)1536 * 384 * 2;
  _Float16* swo0T = (_Float16*)(w + off);  off += (size_t)384 * 768 * 2;
  _Float16* swi1T = (_Float16*)(w + off);  off += (size_t)1536 * 384 * 2;
  _Float16* swo1T = (_Float16*)(w + off);  off += (size_t)384 * 768 * 2;
  _Float16* wpT   = (_Float16*)(w + off);  off += (size_t)128 * 256 * 2;
  _Float16* pwi0T = (_Float16*)(w + off);  off += (size_t)512 * 128 * 2;
  _Float16* pwo0T = (_Float16*)(w + off);  off += (size_t)128 * 256 * 2;
  _Float16* pwi1T = (_Float16*)(w + off);  off += (size_t)512 * 128 * 2;
  _Float16* pwo1T = (_Float16*)(w + off);  off += (size_t)128 * 256 * 2;

  auto T = [&](const float* src, _Float16* dst, int K, int N, int KP) {
    int total = N * KP;
    k_transpose_f16<<<(total + 255) / 256, 256, 0, stream>>>(src, dst, K, N, KP);
  };
  T(w_single, wsT,  833, 384, 864);
  T(swi0,     swi0T, 384, 1536, 384);
  T(swo0,     swo0T, 768, 384, 768);
  T(swi1,     swi1T, 384, 1536, 384);
  T(swo1,     swo1T, 768, 384, 768);
  T(w_pair,   wpT,   256, 128, 256);
  T(pwi0,     pwi0T, 128, 512, 128);
  T(pwo0,     pwo0T, 256, 128, 256);
  T(pwi1,     pwi1T, 128, 512, 128);
  T(pwo1,     pwo1T, 256, 128, 256);

  k_noise_embed<<<1, 256, 0, stream>>>(noise, fourier_w, fourier_b, ln_noise_g,
                                       w_noise, n_buf);
  k_single_s0<<<48, 256, 0, stream>>>(tsc, inputs, ln_single_g, wsT, s0);

  float* out_s = (float*)d_out;                  // (4,768,384)
  float* out_x = out_s + (size_t)4 * 768 * 384;  // (768,768,128)

  k_single_ffn<<<192, 256, 0, stream>>>(s0, n_buf, sg0, swi0T, swo0T,
                                        sg1, swi1T, swo1T, out_s);

  k_pair<<<768 * 24, 256, 0, stream>>>(seq_index, seq_color, seq_sym, seq_entity,
                                       tpc, w_relpos, ln_pair_g, wpT,
                                       pg0, pwi0T, pwo0T, pg1, pwi1T, pwo1T, out_x);
}
